// QuantizedMBConv_34248069218736
// MI455X (gfx1250) — matrix operand order repaired and checked by reading, not compiled, not run on hardware
//
#include <hip/hip_runtime.h>
#include <hip/hip_fp16.h>

typedef __attribute__((ext_vector_type(16))) _Float16 v16h;
typedef __attribute__((ext_vector_type(8)))  _Float16 v8h;
typedef __attribute__((ext_vector_type(8)))  float    v8f;
typedef __attribute__((ext_vector_type(4)))  unsigned int v4u;
typedef __attribute__((ext_vector_type(8)))  int v8i;
typedef __attribute__((ext_vector_type(4)))  int v4i;

#define CIN   96
#define HIDC  576
#define REDC  24
#define NBATCH 32
#define WDIM  56
#define HWSZ  (WDIM*WDIM)          // 3136
#define MTOT  (NBATCH*HWSZ)        // 100352
#define QMAX  127.0f

__device__ __forceinline__ float silu_f(float v) { return v / (1.0f + __expf(-v)); }

// LDS byte offset of a __shared__ object: generic pointers to LDS carry the
// AS(3) offset in their low 32 bits (aperture base occupies the high dword).
__device__ __forceinline__ unsigned lds_off_of(const void* p) {
  return (unsigned)(unsigned long long)p;
}

// ---------------------------------------------------------------------------
// prep: per-tensor quant scales (max|w|/127) + folded BN params
// ---------------------------------------------------------------------------
__global__ __launch_bounds__(256) void prep_kernel(
    const float* __restrict__ w1, const float* __restrict__ w2, const float* __restrict__ w3,
    const float* __restrict__ g1, const float* __restrict__ b1, const float* __restrict__ m1, const float* __restrict__ v1,
    const float* __restrict__ g2, const float* __restrict__ b2, const float* __restrict__ m2, const float* __restrict__ v2,
    const float* __restrict__ g3, const float* __restrict__ b3, const float* __restrict__ m3, const float* __restrict__ v3,
    float* __restrict__ scales,
    float* __restrict__ inv1, float* __restrict__ bet1,
    float* __restrict__ inv2, float* __restrict__ bet2,
    float* __restrict__ inv3, float* __restrict__ bet3)
{
  __shared__ float red[256];
  const int which = blockIdx.x;
  const float* w = (which == 0) ? w1 : (which == 1) ? w2 : w3;
  const int n = (which == 1) ? HIDC * 9 : HIDC * CIN;
  float mx = 0.0f;
  for (int i = threadIdx.x; i < n; i += 256) mx = fmaxf(mx, fabsf(w[i]));
  red[threadIdx.x] = mx;
  __syncthreads();
  for (int s = 128; s > 0; s >>= 1) {
    if (threadIdx.x < s) red[threadIdx.x] = fmaxf(red[threadIdx.x], red[threadIdx.x + s]);
    __syncthreads();
  }
  if (threadIdx.x == 0) scales[which] = red[0] / QMAX;

  const float *g, *bb, *mm, *vv;
  float *pi, *pb;
  int cn;
  if (which == 0)      { g = g1; bb = b1; mm = m1; vv = v1; pi = inv1; pb = bet1; cn = HIDC; }
  else if (which == 1) { g = g2; bb = b2; mm = m2; vv = v2; pi = inv2; pb = bet2; cn = HIDC; }
  else                 { g = g3; bb = b3; mm = m3; vv = v3; pi = inv3; pb = bet3; cn = CIN;  }
  for (int i = threadIdx.x; i < cn; i += 256) {
    float inv = g[i] / sqrtf(vv[i] + 1e-5f);
    pi[i] = inv;
    pb[i] = bb[i] - mm[i] * inv;
  }
}

// ---------------------------------------------------------------------------
// expand: GEMM [M=100352,K=96]x[96,N=576] via v_wmma_f32_16x16x32_f16
// ---------------------------------------------------------------------------
#define EXP_BM 128
#define EXP_BN 64
#define AS_STR 104

__global__ __launch_bounds__(256) void expand_kernel(
    const float* __restrict__ x, const float* __restrict__ w1,
    const float* __restrict__ scales,
    const float* __restrict__ inv1, const float* __restrict__ bet1,
    _Float16* __restrict__ h1)
{
  __shared__ __align__(32) _Float16 As[EXP_BM * AS_STR];
  __shared__ __align__(32) _Float16 Bt[EXP_BN * CIN];
  const int tid  = threadIdx.x;
  const int wave = tid >> 5, lane = tid & 31;
  const int l = lane & 15, hi = lane >> 4;
  const int m0 = blockIdx.x * EXP_BM;
  const int n0 = blockIdx.y * EXP_BN;
  const float s1 = scales[0];

  for (int i = tid; i < CIN * EXP_BM; i += 256) {
    int mi = i & (EXP_BM - 1), k = i >> 7;
    int m = m0 + mi;
    int b = m / HWSZ, hw = m % HWSZ;
    As[mi * AS_STR + k] = (_Float16)x[((size_t)b * CIN + k) * HWSZ + hw];
  }
  for (int i = tid; i < EXP_BN * CIN; i += 256) {
    int k = i % CIN, n = i / CIN;
    float w = w1[(size_t)(n0 + n) * CIN + k];
    Bt[n * CIN + k] = (_Float16)(rintf(w / s1) * s1);
  }
  __syncthreads();

  v8f acc[4] = {};
  const _Float16* arow = As + (wave * 16 + l) * AS_STR;
#pragma unroll
  for (int kc = 0; kc < CIN; kc += 32) {
    union { v16h v; v8h h[2]; } a;
    a.h[0] = *(const v8h*)(arow + kc + hi * 16);
    a.h[1] = *(const v8h*)(arow + kc + hi * 16 + 8);
#pragma unroll
    for (int t = 0; t < 4; ++t) {
      v16h bf = *(const v16h*)(Bt + (t * 16 + l) * CIN + kc + hi * 16);
      acc[t] = __builtin_amdgcn_wmma_f32_16x16x32_f16(
          false, a.v, false, bf, (short)0, acc[t], false, false);
    }
  }

#pragma unroll
  for (int t = 0; t < 4; ++t) {
    int n = n0 + t * 16 + l;
    float inv = inv1[n], bet = bet1[n];
#pragma unroll
    for (int r = 0; r < 8; ++r) {
      int m = m0 + wave * 16 + hi * 8 + r;
      h1[(size_t)m * HIDC + n] = (_Float16)silu_f(acc[t][r] * inv + bet);
    }
  }
}

// ---------------------------------------------------------------------------
// depthwise 3x3: LDS halo tile 16x16x64ch staged by the Tensor Data Mover
// (interior tiles: one 3D tensor_load_to_lds, TENSORcnt) or per-lane
// global_load_async_to_lds_b128 with ds zero-fill (boundary tiles, ASYNCcnt).
// ---------------------------------------------------------------------------
__global__ __launch_bounds__(256) void dw_kernel(
    const _Float16* __restrict__ h1, const float* __restrict__ w2,
    const float* __restrict__ scales,
    const float* __restrict__ inv2, const float* __restrict__ bet2,
    _Float16* __restrict__ h2)
{
  __shared__ __align__(128) _Float16 T[16 * 16 * 64];   // 32 KB, channel-fastest
  __shared__ float Wq[64 * 9];
  __shared__ float inv_s[64], bet_s[64];
  const float s2 = scales[1];
  const int b = blockIdx.z;
  const int cg = blockIdx.y * 64;
  const int tile = blockIdx.x;
  const int tr = tile >> 2, tc = tile & 3;
  const int th0 = tr * 14 - 1;
  const int tw0 = tc * 14 - 1;
  const bool interior = (tr >= 1) && (tr <= 2) && (tc >= 1) && (tc <= 2);

  for (int i = threadIdx.x; i < 64 * 9; i += 256) {
    int c = i / 9;
    float w = w2[(size_t)(cg + c) * 9 + (i % 9)];
    Wq[i] = rintf(w / s2) * s2;
  }
  if (threadIdx.x < 64) {
    inv_s[threadIdx.x] = inv2[cg + threadIdx.x];
    bet_s[threadIdx.x] = bet2[cg + threadIdx.x];
  }

  if (interior) {
    // ---- TDM path: 3D tile dim0=64ch(contig), dim1=16 cols, dim2=16 rows ----
    if (threadIdx.x < 32) {
      unsigned long long ga = (unsigned long long)(size_t)
          &h1[((size_t)b * HWSZ + th0 * WDIM + tw0) * HIDC + cg];
      unsigned lds = lds_off_of(&T[0]);
      v4u g0;
      g0.x = 1u;                                   // count=1, user desc
      g0.y = lds;                                  // lds_addr
      g0.z = (unsigned)(ga & 0xffffffffu);         // global_addr[31:0]
      g0.w = (unsigned)((ga >> 32) & 0x01ffffffu)  // global_addr[56:32]
             | (2u << 30);                         // type=2 (image)
      v8i g1v;
      g1v[0] = (int)(1u << 16);                    // wg_mask=0, data_size=2B
      g1v[1] = (int)(64u << 16);                   // tensor_dim0=64 (lo16 @ [31:16])
      g1v[2] = (int)((64u >> 16) | ((unsigned)WDIM << 16));   // td0 hi / tensor_dim1=56 lo
      g1v[3] = (int)(64u << 16);                   // td1 hi=0 | tile_dim0=64
      g1v[4] = (int)(16u | (16u << 16));           // tile_dim1=16 | tile_dim2=16
      g1v[5] = (int)(unsigned)HIDC;                // tensor_dim0_stride lo32 = 576
      g1v[6] = (int)(((unsigned)(HIDC * WDIM) & 0xffffu) << 16); // td0s hi | td1 stride lo16
      g1v[7] = (int)((unsigned)(HIDC * WDIM) >> 16);            // td1 stride hi32
      v4i g2v; g2v[0] = WDIM; g2v[1] = 0; g2v[2] = 0; g2v[3] = 0;  // tensor_dim2=56
      v4i g3v; g3v[0] = 0; g3v[1] = 0; g3v[2] = 0; g3v[3] = 0;
#if __clang_major__ >= 23
      v8i g4v; g4v[0]=0; g4v[1]=0; g4v[2]=0; g4v[3]=0; g4v[4]=0; g4v[5]=0; g4v[6]=0; g4v[7]=0;
      __builtin_amdgcn_tensor_load_to_lds(g0, g1v, g2v, g3v, g4v, 0);
#else
      __builtin_amdgcn_tensor_load_to_lds(g0, g1v, g2v, g3v, 0);
#endif
      __builtin_amdgcn_s_wait_tensorcnt(0);
    }
  } else {
    // ---- async path: 16B chunks, 8 per (row,col); zero-fill OOB via DS ----
    for (int i = threadIdx.x; i < 16 * 16 * 8; i += 256) {
      int ch = i & 7;            // 16-byte chunk within the 128B channel row
      int col = (i >> 3) & 15;
      int row = i >> 7;
      int hh = th0 + row, ww = tw0 + col;
      _Float16* dst = &T[((row * 16) + col) * 64 + ch * 8];
      if (hh >= 0 && hh < WDIM && ww >= 0 && ww < WDIM) {
        unsigned lds = lds_off_of(dst);
        unsigned long long ga = (unsigned long long)(size_t)
            &h1[((size_t)b * HWSZ + hh * WDIM + ww) * HIDC + cg + ch * 8];
        asm volatile("global_load_async_to_lds_b128 %0, %1, off"
                     :: "v"(lds), "v"(ga) : "memory");
      } else {
        *(v8h*)dst = (v8h)(_Float16)0.0f;
      }
    }
    asm volatile("s_wait_asynccnt 0x0" ::: "memory");
  }
  __syncthreads();

  for (int i = threadIdx.x; i < 14 * 14 * 64; i += 256) {
    int c = i & 63, p = i >> 6;
    int col = p % 14, row = p / 14;
    float acc = 0.0f;
#pragma unroll
    for (int ky = 0; ky < 3; ++ky)
#pragma unroll
      for (int kx = 0; kx < 3; ++kx)
        acc += (float)T[((row + ky) * 16 + (col + kx)) * 64 + c] * Wq[c * 9 + ky * 3 + kx];
    float v = silu_f(acc * inv_s[c] + bet_s[c]);
    int hh = th0 + 1 + row, ww = tw0 + 1 + col;
    h2[((size_t)b * HWSZ + hh * WDIM + ww) * HIDC + cg + c] = (_Float16)v;
  }
}

// ---------------------------------------------------------------------------
// global average pool over HW per (b, c)
// ---------------------------------------------------------------------------
__global__ __launch_bounds__(256) void pool_kernel(
    const _Float16* __restrict__ h2, float* __restrict__ pooled)
{
  const int b = blockIdx.x;
  for (int c = threadIdx.x; c < HIDC; c += 256) {
    float s = 0.0f;
    for (int hw = 0; hw < HWSZ; ++hw)
      s += (float)h2[((size_t)b * HWSZ + hw) * HIDC + c];
    pooled[b * HIDC + c] = s * (1.0f / HWSZ);
  }
}

// ---------------------------------------------------------------------------
// SE gate
// ---------------------------------------------------------------------------
__global__ __launch_bounds__(256) void se_kernel(
    const float* __restrict__ pooled,
    const float* __restrict__ sew1, const float* __restrict__ seb1,
    const float* __restrict__ sew2, const float* __restrict__ seb2,
    float* __restrict__ sgate)
{
  __shared__ float t1[NBATCH * REDC];
  for (int i = threadIdx.x; i < NBATCH * REDC; i += 256) {
    int b = i / REDC, r = i % REDC;
    float a = seb1[r];
    for (int c = 0; c < HIDC; ++c) a += pooled[b * HIDC + c] * sew1[r * HIDC + c];
    t1[i] = silu_f(a);
  }
  __syncthreads();
  for (int i = threadIdx.x; i < NBATCH * HIDC; i += 256) {
    int b = i / HIDC, c = i % HIDC;
    float a = seb2[c];
#pragma unroll
    for (int r = 0; r < REDC; ++r) a += t1[b * REDC + r] * sew2[c * REDC + r];
    sgate[i] = 1.0f / (1.0f + __expf(-a));
  }
}

// ---------------------------------------------------------------------------
// project: GEMM [M=100352,K=576]x[576,N=96] via WMMA + SE gate + BN + residual
// ---------------------------------------------------------------------------
#define PRJ_BM 128
#define PAS_STR 40

__global__ __launch_bounds__(256) void project_kernel(
    const _Float16* __restrict__ h2, const float* __restrict__ sgate,
    const float* __restrict__ w3, const float* __restrict__ scales,
    const float* __restrict__ inv3, const float* __restrict__ bet3,
    const float* __restrict__ x, float* __restrict__ out)
{
  __shared__ __align__(32) _Float16 As[PRJ_BM * PAS_STR];
  __shared__ __align__(32) _Float16 Bt[CIN * 32];
  const int tid  = threadIdx.x;
  const int wave = tid >> 5, lane = tid & 31;
  const int l = lane & 15, hi = lane >> 4;
  const int m0 = blockIdx.x * PRJ_BM;
  const float s3 = scales[2];

  v8f acc[6] = {};
  for (int kc = 0; kc < HIDC; kc += 32) {
    __syncthreads();
    for (int i = tid; i < PRJ_BM * 32; i += 256) {
      int k = i & 31, mi = i >> 5;
      int m = m0 + mi;
      int b = m / HWSZ;
      float a = (float)h2[(size_t)m * HIDC + kc + k] * sgate[b * HIDC + kc + k];
      As[mi * PAS_STR + k] = (_Float16)a;
    }
    for (int i = tid; i < CIN * 32; i += 256) {
      int k = i & 31, n = i >> 5;
      float w = w3[(size_t)n * HIDC + kc + k];
      Bt[n * 32 + k] = (_Float16)(rintf(w / s3) * s3);
    }
    if (kc + 32 < HIDC) {
      int mi = tid & (PRJ_BM - 1);
      __builtin_prefetch(&h2[(size_t)(m0 + mi) * HIDC + kc + 32], 0, 1);
    }
    __syncthreads();

    union { v16h v; v8h h[2]; } a;
    const _Float16* arow = As + (wave * 16 + l) * PAS_STR;
    a.h[0] = *(const v8h*)(arow + hi * 16);
    a.h[1] = *(const v8h*)(arow + hi * 16 + 8);
#pragma unroll
    for (int t = 0; t < 6; ++t) {
      v16h bf = *(const v16h*)(Bt + (t * 16 + l) * 32 + hi * 16);
      acc[t] = __builtin_amdgcn_wmma_f32_16x16x32_f16(
          false, a.v, false, bf, (short)0, acc[t], false, false);
    }
  }

#pragma unroll
  for (int t = 0; t < 6; ++t) {
    int n = t * 16 + l;
    float inv = inv3[n], bet = bet3[n];
#pragma unroll
    for (int r = 0; r < 8; ++r) {
      int m = m0 + wave * 16 + hi * 8 + r;
      int b = m / HWSZ, hw = m % HWSZ;
      size_t idx = ((size_t)b * CIN + n) * HWSZ + hw;
      out[idx] = acc[t][r] * inv + bet + x[idx];
    }
  }
}

// ---------------------------------------------------------------------------
extern "C" void kernel_launch(void* const* d_in, const int* in_sizes, int n_in,
                              void* d_out, int out_size, void* d_ws, size_t ws_size,
                              hipStream_t stream)
{
  const float* x    = (const float*)d_in[0];
  const float* w1   = (const float*)d_in[1];
  const float* g1   = (const float*)d_in[2];
  const float* b1   = (const float*)d_in[3];
  const float* m1   = (const float*)d_in[4];
  const float* v1   = (const float*)d_in[5];
  const float* w2   = (const float*)d_in[6];
  const float* g2   = (const float*)d_in[7];
  const float* b2   = (const float*)d_in[8];
  const float* m2   = (const float*)d_in[9];
  const float* v2   = (const float*)d_in[10];
  const float* sew1 = (const float*)d_in[11];
  const float* seb1 = (const float*)d_in[12];
  const float* sew2 = (const float*)d_in[13];
  const float* seb2 = (const float*)d_in[14];
  const float* w3   = (const float*)d_in[15];
  const float* g3   = (const float*)d_in[16];
  const float* b3   = (const float*)d_in[17];
  const float* m3   = (const float*)d_in[18];
  const float* v3   = (const float*)d_in[19];
  float* out = (float*)d_out;
  (void)in_sizes; (void)n_in; (void)out_size; (void)ws_size;

  char* ws = (char*)d_ws;
  size_t off = 0;
  auto take = [&](size_t bytes) -> char* {
    char* p = ws + off;
    off = (off + bytes + 255) & ~(size_t)255;
    return p;
  };
  float* scales = (float*)take(3 * sizeof(float));
  float* inv1 = (float*)take(HIDC * 4); float* bet1 = (float*)take(HIDC * 4);
  float* inv2 = (float*)take(HIDC * 4); float* bet2 = (float*)take(HIDC * 4);
  float* inv3 = (float*)take(CIN * 4);  float* bet3 = (float*)take(CIN * 4);
  float* pooled = (float*)take((size_t)NBATCH * HIDC * 4);
  float* sgate  = (float*)take((size_t)NBATCH * HIDC * 4);
  _Float16* h1 = (_Float16*)take((size_t)MTOT * HIDC * 2);
  _Float16* h2 = (_Float16*)take((size_t)MTOT * HIDC * 2);

  prep_kernel<<<3, 256, 0, stream>>>(w1, w2, w3,
                                     g1, b1, m1, v1, g2, b2, m2, v2, g3, b3, m3, v3,
                                     scales, inv1, bet1, inv2, bet2, inv3, bet3);
  expand_kernel<<<dim3(MTOT / EXP_BM, HIDC / EXP_BN), 256, 0, stream>>>(
      x, w1, scales, inv1, bet1, h1);
  dw_kernel<<<dim3(16, HIDC / 64, NBATCH), 256, 0, stream>>>(
      h1, w2, scales, inv2, bet2, h2);
  pool_kernel<<<NBATCH, 256, 0, stream>>>(h2, pooled);
  se_kernel<<<1, 256, 0, stream>>>(pooled, sew1, seb1, sew2, seb2, sgate);
  project_kernel<<<MTOT / PRJ_BM, 256, 0, stream>>>(
      h2, sgate, w3, scales, inv3, bet3, x, out);
}
